// CopyDecoder_81449759801666
// MI455X (gfx1250) — matrix operand order repaired and checked by reading, not compiled, not run on hardware
//
#include <hip/hip_runtime.h>
#include <hip/hip_bf16.h>
#include <math.h>
#include <cstdint>

// ---------------------------------------------------------------------------
// CopyDecoder for MI455X (gfx1250, wave32).
// GEMMs: v_wmma_f32_16x16x32_bf16 (bf16 A/B, f32 accum), A-tile staged into
// LDS with global_load_async_to_lds_b128 (ASYNCcnt) and read back via ds_load.
// Shapes: B=64, SEQ=200, EMBED=256, HIDDEN=512, h2=1024, VOCAB=50000.
// ---------------------------------------------------------------------------

#define BB   64
#define SEQ  200
#define EMB  256
#define HID  512
#define H2   1024
#define VOC  50000
#define G3   1536      // 3*HIDDEN
#define XK   1280      // EMBED + h2

typedef __attribute__((ext_vector_type(16))) __bf16 v16bf;
typedef __attribute__((ext_vector_type(8)))  float  v8f;

// ---------------------------------------------------------------------------
// small helpers
// ---------------------------------------------------------------------------
__device__ __forceinline__ float warp_red_sum(float v) {
    #pragma unroll
    for (int o = 16; o; o >>= 1) v += __shfl_down(v, o, 32);
    return v;
}
__device__ __forceinline__ float warp_red_max(float v) {
    #pragma unroll
    for (int o = 16; o; o >>= 1) v = fmaxf(v, __shfl_down(v, o, 32));
    return v;
}
__device__ __forceinline__ float sigmoidf_(float x) {
    return 1.0f / (1.0f + __expf(-x));
}

// ---------------------------------------------------------------------------
// K1: x = concat(embed_table[input_idx[b]], weighted[b])  -> (B, 1280)
// ---------------------------------------------------------------------------
__global__ void build_x_kernel(const int* __restrict__ input_idx,
                               const float* __restrict__ embed,
                               const float* __restrict__ weighted,
                               float* __restrict__ x) {
    int b = blockIdx.x;
    const float* e  = embed + (size_t)input_idx[b] * EMB;
    const float* w  = weighted + (size_t)b * H2;
    float* xb = x + (size_t)b * XK;
    for (int i = threadIdx.x; i < EMB; i += blockDim.x) xb[i] = e[i];
    for (int i = threadIdx.x; i < H2;  i += blockDim.x) xb[EMB + i] = w[i];
}

__global__ void zero_f32_kernel(float* __restrict__ p, int n) {
    int i = blockIdx.x * blockDim.x + threadIdx.x;
    if (i < n) p[i] = 0.0f;
}

// ---------------------------------------------------------------------------
// Generic  C[M,N] = A[M,K] @ W[N,K]^T + bias[N]   via WMMA bf16 (f32 accum).
//
// Block = 256 thr = 8 waves, tiled 2(M) x 4(N) -> 32 x 64 output per block.
// Per K-slab of 64: the 32x64 f32 A tile (8 KB) is staged into LDS with
// global_load_async_to_lds_b128 (one b128 chunk per thread x2), waited with
// s_wait_asynccnt + barrier, then all 4 N-waves of an M-row reuse it (4x
// less global A traffic). B fragments stream straight from global.
// Requires K % 64 == 0, M % 32 == 0, N % 16 == 0. Tail N-waves clamp their
// loads and skip the store (no early return: block has barriers).
// ---------------------------------------------------------------------------
__global__ void gemm_bf16_wmma(const float* __restrict__ A,
                               const float* __restrict__ W,
                               const float* __restrict__ bias,
                               float* __restrict__ C,
                               int M, int N, int K) {
    __shared__ float shA[32 * 64];                 // 8 KB A slab

    const int t    = threadIdx.x;
    const int wave = t >> 5;
    const int lane = t & 31;
    const int m0b  = blockIdx.y * 32;                    // block M base
    const int m0   = m0b + (wave >> 2) * 16;             // wave M base
    const int n0   = blockIdx.x * 64 + (wave & 3) * 16;  // wave N base
    const bool nvalid = (n0 < N);

    const int nrow = nvalid ? (n0 + (lane & 15)) : 0;    // clamped B row
    const int kbA  = (lane >> 4) * 8;                    // A lane K sub-base
    const int kbB  = (lane >> 4) * 8;                    // B lane K sub-base

    const float* __restrict__ wrow = W + (size_t)nrow * K;
    const float* __restrict__ aLds =
        &shA[(size_t)(((wave >> 2) * 16) + (lane & 15)) * 64];

    v8f acc = {};
    for (int kc = 0; kc < K; kc += 64) {
        // ---- stage 32 rows x 64 K of A into LDS (async, 512 b128 chunks) ----
        #pragma unroll
        for (int j = 0; j < 2; ++j) {
            int c  = t + j * 256;            // chunk 0..511
            int r  = c >> 4;                 // tile row 0..31
            int kq = (c & 15) << 2;          // float col 0..60
            const float* gsrc = A + (size_t)(m0b + r) * K + (kc + kq);
            unsigned ldsaddr  = (unsigned)(size_t)(&shA[r * 64 + kq]);
            asm volatile("global_load_async_to_lds_b128 %0, %1, off"
                         :: "v"(ldsaddr),
                            "v"((unsigned long long)(size_t)gsrc)
                         : "memory");
        }
        asm volatile("s_wait_asynccnt 0x0" ::: "memory");
        __syncthreads();

        __builtin_prefetch(wrow + kc + 64, 0, 1);    // global_prefetch_b8

        // ---- two 16x16x32 WMMAs over the 64-wide slab ----
        #pragma unroll
        for (int h = 0; h < 2; ++h) {
            const int ks = h * 32;
            v16bf a, bfr;
            #pragma unroll
            for (int i = 0; i < 8; ++i)  a[i]     = (__bf16)aLds[ks + kbA + i];
            #pragma unroll
            for (int i = 0; i < 8; ++i)  a[8 + i] = (__bf16)aLds[ks + 16 + kbA + i];
            #pragma unroll
            for (int i = 0; i < 8; ++i)  bfr[i]     = (__bf16)wrow[kc + ks + kbB + i];
            #pragma unroll
            for (int i = 0; i < 8; ++i)  bfr[8 + i] = (__bf16)wrow[kc + ks + 16 + kbB + i];
            acc = __builtin_amdgcn_wmma_f32_16x16x32_bf16(
                      false, a, false, bfr, (short)0, acc, false, false);
        }
        __syncthreads();   // protect slab before next stage overwrites it
    }

    if (nvalid) {
        const int mo = m0 + (lane >> 4) * 8;
        const int nc = n0 + (lane & 15);
        const float bv = bias[nc];
        #pragma unroll
        for (int r = 0; r < 8; ++r)
            C[(size_t)(mo + r) * N + nc] = acc[r] + bv;
    }
}

// ---------------------------------------------------------------------------
// K5: GRU gate combine:  state = (1-z)*n + z*h
// ---------------------------------------------------------------------------
__global__ void gru_combine_kernel(const float* __restrict__ gx,
                                   const float* __restrict__ gh,
                                   const float* __restrict__ h,
                                   float* __restrict__ state) {
    int i = blockIdx.x * blockDim.x + threadIdx.x;   // 0 .. B*HID-1
    if (i >= BB * HID) return;
    int b = i >> 9, j = i & (HID - 1);
    const float* gxb = gx + (size_t)b * G3;
    const float* ghb = gh + (size_t)b * G3;
    float r = sigmoidf_(gxb[j]           + ghb[j]);
    float z = sigmoidf_(gxb[HID + j]     + ghb[HID + j]);
    float n = tanhf   (gxb[2 * HID + j] + r * ghb[2 * HID + j]);
    state[i] = (1.0f - z) * n + z * h[i];
}

// ---------------------------------------------------------------------------
// K7: attention score, fused:
//   sch = tanh(encoded(12800x1024) @ wc_w(512x1024)^T + wc_b)   [WMMA]
//   score_c[row] += sum_n sch[row][n] * state[b][n]              [epilogue]
// Epilogue: half-wave shuffle reduce over the 16 lanes that share a row,
// then one global_atomic_add_f32 per (row, n-tile). score_c pre-zeroed.
// ---------------------------------------------------------------------------
__global__ void attn_score_gemm(const float* __restrict__ enc,
                                const float* __restrict__ wc_w,
                                const float* __restrict__ wc_b,
                                const float* __restrict__ state,
                                float* __restrict__ score_c) {
    const int K = H2;
    const int wave = threadIdx.x >> 5;
    const int lane = threadIdx.x & 31;
    const int m0 = blockIdx.y * 32 + (wave >> 2) * 16;   // bs-row tile
    const int n0 = blockIdx.x * 64 + (wave & 3) * 16;

    const int mrow = m0 + (lane & 15);
    const int nrow = n0 + (lane & 15);
    const int kbA  = (lane >> 4) * 8;
    const int kbB  = (lane >> 4) * 8;

    const float* __restrict__ arow = enc  + (size_t)mrow * K;
    const float* __restrict__ wrow = wc_w + (size_t)nrow * K;

    v8f acc = {};
    #pragma unroll 2
    for (int kc = 0; kc < K; kc += 32) {
        __builtin_prefetch(arow + kc + 64, 0, 1);
        v16bf a, bfr;
        #pragma unroll
        for (int i = 0; i < 8; ++i)  a[i]     = (__bf16)arow[kc + kbA + i];
        #pragma unroll
        for (int i = 0; i < 8; ++i)  a[8 + i] = (__bf16)arow[kc + 16 + kbA + i];
        #pragma unroll
        for (int i = 0; i < 8; ++i)  bfr[i]     = (__bf16)wrow[kc + kbB + i];
        #pragma unroll
        for (int i = 0; i < 8; ++i)  bfr[8 + i] = (__bf16)wrow[kc + 16 + kbB + i];
        acc = __builtin_amdgcn_wmma_f32_16x16x32_bf16(
                  false, a, false, bfr, (short)0, acc, false, false);
    }

    const int mo = m0 + (lane >> 4) * 8;
    const int nc = n0 + (lane & 15);
    const float bv = wc_b[nc];
    #pragma unroll
    for (int r = 0; r < 8; ++r) {
        int row = mo + r;                    // 0..12799 (= b*SEQ + s)
        int b   = row / SEQ;
        float v = tanhf(acc[r] + bv) * state[(size_t)b * HID + nc];
        // sum across the 16 lanes that hold this row (same half-wave)
        #pragma unroll
        for (int o = 8; o; o >>= 1) v += __shfl_down(v, o, 16);
        if ((lane & 15) == 0) atomicAdd(&score_c[row], v);
    }
}

// ---------------------------------------------------------------------------
// K8: per-batch-row softmax over [score_g | masked score_c] (50200 logits),
// write prob_g to out, scatter-add prob_c at encoded_idx, build attn,
// normalize, and compute weighted_new = attn . encoded.  One block per b.
// ---------------------------------------------------------------------------
__global__ void softmax_out_kernel(const float* __restrict__ score_g,
                                   const float* __restrict__ score_c,
                                   const int* __restrict__ enc_idx,
                                   const int* __restrict__ input_idx,
                                   const float* __restrict__ encoded,
                                   float* __restrict__ out,
                                   float* __restrict__ weighted_new,
                                   float* __restrict__ prob_c) {
    __shared__ float red[8];
    __shared__ float attn_sh[SEQ];
    __shared__ float sh_max, sh_sum, sh_asum;

    const int b = blockIdx.x, t = threadIdx.x;
    const float* __restrict__ sg = score_g + (size_t)b * VOC;
    const int*   __restrict__ ei = enc_idx + b * SEQ;
    const int inp = input_idx[b];

    // ---- pass 1: max over both segments (masked score_c) ----
    float mx = -3.4e38f;
    for (int i = t; i < VOC; i += 256) mx = fmaxf(mx, sg[i]);
    for (int i = t; i < SEQ; i += 256) {
        int e = ei[i];
        float v = (e != 0 && e == inp) ? score_c[b * SEQ + i] : 0.0f;
        mx = fmaxf(mx, v);
    }
    mx = warp_red_max(mx);
    if ((t & 31) == 0) red[t >> 5] = mx;
    __syncthreads();
    if (t == 0) { float m = red[0]; for (int i = 1; i < 8; ++i) m = fmaxf(m, red[i]); sh_max = m; }
    __syncthreads();
    mx = sh_max;

    // ---- pass 2: sum of exp ----
    float sm = 0.0f;
    for (int i = t; i < VOC; i += 256) sm += __expf(sg[i] - mx);
    for (int i = t; i < SEQ; i += 256) {
        int e = ei[i];
        float v = (e != 0 && e == inp) ? score_c[b * SEQ + i] : 0.0f;
        sm += __expf(v - mx);
    }
    sm = warp_red_sum(sm);
    __syncthreads();
    if ((t & 31) == 0) red[t >> 5] = sm;
    __syncthreads();
    if (t == 0) { float s = 0; for (int i = 0; i < 8; ++i) s += red[i]; sh_sum = s; }
    __syncthreads();
    const float inv = 1.0f / sh_sum;

    // ---- pass 3: prob_g -> out, prob_c -> ws, attn (pre-scale) -> LDS ----
    for (int i = t; i < VOC; i += 256)
        out[(size_t)b * VOC + i] = __expf(sg[i] - mx) * inv;
    for (int i = t; i < SEQ; i += 256) {
        int e = ei[i];
        float v = (e != 0 && e == inp) ? score_c[b * SEQ + i] : 0.0f;
        float p = __expf(v - mx) * inv;
        prob_c[b * SEQ + i] = p;
        attn_sh[i] = (e == inp) ? p : 0.0f;   // idx_from_input mask only
    }
    __syncthreads();

    // ---- scatter prob_c into out (indices may repeat -> atomic) ----
    for (int i = t; i < SEQ; i += 256)
        atomicAdd(&out[(size_t)b * VOC + ei[i]], prob_c[b * SEQ + i]);

    // ---- attn normalization sum ----
    float as = 0.0f;
    for (int i = t; i < SEQ; i += 256) as += attn_sh[i];
    as = warp_red_sum(as);
    __syncthreads();
    if ((t & 31) == 0) red[t >> 5] = as;
    __syncthreads();
    if (t == 0) { float s = 0; for (int i = 0; i < 8; ++i) s += red[i]; sh_asum = s; }
    __syncthreads();
    const float scale = (sh_asum > 0.0f) ? 1.0f / sh_asum : 1.0f;

    // ---- weighted_new[b][k] = sum_s attn[s] * encoded[b][s][k] ----
    const float* __restrict__ eb = encoded + (size_t)b * SEQ * H2;
    for (int k = t; k < H2; k += 256) {
        float acc = 0.0f;
        for (int s = 0; s < SEQ; ++s)
            acc += attn_sh[s] * eb[(size_t)s * H2 + k];
        weighted_new[(size_t)b * H2 + k] = acc * scale;
    }
}

// ---------------------------------------------------------------------------
// launcher
// ---------------------------------------------------------------------------
extern "C" void kernel_launch(void* const* d_in, const int* in_sizes, int n_in,
                              void* d_out, int out_size, void* d_ws, size_t ws_size,
                              hipStream_t stream) {
    (void)in_sizes; (void)n_in; (void)out_size; (void)ws_size;

    const int*   input_idx   = (const int*)  d_in[0];
    const float* encoded     = (const float*)d_in[1];
    const int*   encoded_idx = (const int*)  d_in[2];
    const float* prev_state  = (const float*)d_in[3];
    const float* weighted    = (const float*)d_in[4];
    // d_in[5] = order (unused by the reference math)
    const float* embed       = (const float*)d_in[6];
    const float* w_ih        = (const float*)d_in[7];
    const float* w_hh        = (const float*)d_in[8];
    const float* b_ih        = (const float*)d_in[9];
    const float* b_hh        = (const float*)d_in[10];
    const float* wo_w        = (const float*)d_in[11];
    const float* wo_b        = (const float*)d_in[12];
    const float* wc_w        = (const float*)d_in[13];
    const float* wc_b        = (const float*)d_in[14];

    float* out          = (float*)d_out;                    // (B, 1, VOC)
    float* state        = out + (size_t)BB * VOC;           // (B, HID)
    float* weighted_new = state + (size_t)BB * HID;         // (B, 1, H2)

    float* ws      = (float*)d_ws;
    float* x       = ws;                                    // B*1280
    float* gx      = x       + (size_t)BB * XK;             // B*1536
    float* gh      = gx      + (size_t)BB * G3;             // B*1536
    float* score_g = gh      + (size_t)BB * G3;             // B*VOC
    float* score_c = score_g + (size_t)BB * VOC;            // B*SEQ
    float* prob_c  = score_c + (size_t)BB * SEQ;            // B*SEQ
    // total ws: ~3.50M floats = ~14.0 MB

    // K1: x = [embed[idx], weighted]
    build_x_kernel<<<BB, 256, 0, stream>>>(input_idx, embed, weighted, x);
    // K2: zero score_c accumulators
    zero_f32_kernel<<<(BB * SEQ + 255) / 256, 256, 0, stream>>>(score_c, BB * SEQ);
    // K3: gx = x @ w_ih^T + b_ih            (64 x 1536 x 1280)
    gemm_bf16_wmma<<<dim3(G3 / 64, BB / 32), 256, 0, stream>>>(
        x, w_ih, b_ih, gx, BB, G3, XK);
    // K4: gh = prev_state @ w_hh^T + b_hh   (64 x 1536 x 512)
    gemm_bf16_wmma<<<dim3(G3 / 64, BB / 32), 256, 0, stream>>>(
        prev_state, w_hh, b_hh, gh, BB, G3, HID);
    // K5: GRU combine -> state
    gru_combine_kernel<<<(BB * HID + 255) / 256, 256, 0, stream>>>(
        gx, gh, prev_state, state);
    // K6: score_g = state @ wo_w^T + wo_b   (64 x 50000 x 512)
    gemm_bf16_wmma<<<dim3((VOC + 63) / 64, BB / 32), 256, 0, stream>>>(
        state, wo_w, wo_b, score_g, BB, VOC, HID);
    // K7: fused tanh-GEMM + dot(state) -> score_c  (12800 x 512 x 1024)
    attn_score_gemm<<<dim3(HID / 64, (BB * SEQ) / 32), 256, 0, stream>>>(
        encoded, wc_w, wc_b, state, score_c);
    // K8: softmax + scatter + attention-weighted context
    softmax_out_kernel<<<BB, 256, 0, stream>>>(
        score_g, score_c, encoded_idx, input_idx, encoded,
        out, weighted_new, prob_c);
}
